// MultiChannelGCN_48808008352174
// MI455X (gfx1250) — compile-verified, matrix operand order checked by reading, not run on hardware
//
#include <hip/hip_runtime.h>
#include <hip/hip_bf16.h>

// ---------------------------------------------------------------------------
// MultiChannelGCN on gfx1250 (MI455X):
//   h[c]  = x[c] @ W[c]            -> bf16-split WMMA (3x v_wmma_f32_16x16x32_bf16)
//   out   = b + dinv^2 * h (self-loop) + scatter_add(norm * h[row]) over edges
//
// GEMM staging uses a fragment-native (bit3<->bit4 permuted K) LDS layout so
// every WMMA fragment is two contiguous ds_load_b128 per lane, no element
// packing.
// ---------------------------------------------------------------------------

#define NNODES 10000
#define FDIM   256
#define CH     10
#define EDGES  160000

typedef __attribute__((ext_vector_type(8)))  float  v8f;
typedef __attribute__((ext_vector_type(16))) __bf16 v16bf;

struct XPtrs { const float* p[CH]; };

union FragU { v16bf v; uint4 q[2]; };   // 16 bf16 = 32 B = 2 x b128
union Pk4   { __bf16 b[4]; uint2 u; };  // 4 bf16 = 8 B  = 1 x b64

// K-permutation matching the ISA 16-bit A/B fragment order:
// lanes 0-15 hold K {0..7,16..23}, lanes 16-31 hold K {8..15,24..31},
// i.e. position = k with bits 3 and 4 swapped.
__device__ __forceinline__ int permk(int k) {
    return (k & 7) | ((k & 8) << 1) | ((k & 16) >> 1);
}

// ---------------------------------------------------------------- degree ----
__global__ __launch_bounds__(256) void k_init_deg(float* __restrict__ deg) {
    int i = blockIdx.x * 256 + threadIdx.x;
    if (i < NNODES) deg[i] = 1.0f;                 // self-loop weight
}

__global__ __launch_bounds__(256) void k_deg(const int* __restrict__ ei,
                                             const float* __restrict__ ew,
                                             float* __restrict__ deg) {
    int e = blockIdx.x * 256 + threadIdx.x;
    if (e < EDGES) atomicAdd(&deg[ei[EDGES + e]], ew[e]);   // target = col
}

__global__ __launch_bounds__(256) void k_dinv(const float* __restrict__ deg,
                                              float* __restrict__ dinv) {
    int i = blockIdx.x * 256 + threadIdx.x;
    if (i < NNODES) {
        float d = deg[i];
        dinv[i] = (d > 0.0f) ? rsqrtf(d) : 0.0f;
    }
}

__global__ __launch_bounds__(256) void k_norm(const int* __restrict__ ei,
                                              const float* __restrict__ ew,
                                              const float* __restrict__ dinv,
                                              float* __restrict__ nrm) {
    int e = blockIdx.x * 256 + threadIdx.x;
    if (e < EDGES) nrm[e] = dinv[ei[e]] * ew[e] * dinv[ei[EDGES + e]];
}

// ---------------------------------------------------------------- GEMM ------
// Block: 256 threads = 8 wave32 waves. Block tile 64(M) x 64(N), wave tile
// 16(M) x 32(N) (two 16x16 accumulators, A-fragment reuse). K stepped by 32.
// fp32 -> bf16 hi/lo split in LDS; acc += Ahi*Bhi + Alo*Bhi + Ahi*Blo.
__global__ __launch_bounds__(256) void k_gemm(XPtrs xp,
                                              const float* __restrict__ Wm,
                                              float* __restrict__ h) {
    const int c    = blockIdx.z;
    const int row0 = blockIdx.x * 64;
    const int col0 = blockIdx.y * 64;

    const float* __restrict__ x  = xp.p[c];
    const float* __restrict__ Wc = Wm + (size_t)c * FDIM * FDIM;
    float* __restrict__       hc = h  + (size_t)c * NNODES * FDIM;

    // Row stride 40 bf16 = 80 B: keeps every 16-element fragment chunk 16 B
    // aligned for ds_load_b128 and staggers bank starts.
    __shared__ __attribute__((aligned(16))) __bf16 sAh[64][40]; // A tile hi [m][perm(k)]
    __shared__ __attribute__((aligned(16))) __bf16 sAl[64][40]; // A tile lo
    __shared__ __attribute__((aligned(16))) __bf16 sBh[64][40]; // B tile hi [n][perm(k)]
    __shared__ __attribute__((aligned(16))) __bf16 sBl[64][40]; // B tile lo

    const int tid  = threadIdx.x;
    const int wave = tid >> 5;
    const int lane = tid & 31;
    const int lm   = lane & 15;      // M (A) or N (B/C) index within tile
    const int hb   = lane >> 4;      // K-half / row-half selector
    const int m0   = (wave >> 1) * 16;   // wave row   (0,16,32,48)
    const int n0   = (wave & 1) * 32;    // wave col   (0,32)

    v8f acc0 = {};
    v8f acc1 = {};

    for (int kk = 0; kk < FDIM; kk += 32) {
        // ---- stage A tile (64 x 32 fp32 -> bf16 hi/lo, perm'd K) ----------
#pragma unroll
        for (int it = 0; it < 2; ++it) {
            int p    = tid + it * 256;        // 0..511 float4 slots
            int r    = p >> 3;                // m local: 0..63
            int c4   = (p & 7) * 4;           // k local: 0..28, mult of 4
            int grow = row0 + r;
            float4 v4 = make_float4(0.f, 0.f, 0.f, 0.f);
            if (grow < NNODES) {
                const float* ap = x + (size_t)grow * FDIM + kk + c4;
                v4 = *(const float4*)ap;
                if (kk + 32 < FDIM) __builtin_prefetch(ap + 32, 0, 1); // global_prefetch_b8
            }
            float f[4] = {v4.x, v4.y, v4.z, v4.w};
            Pk4 ph, pl;
#pragma unroll
            for (int j = 0; j < 4; ++j) {
                __bf16 hi = (__bf16)f[j];
                ph.b[j] = hi;
                pl.b[j] = (__bf16)(f[j] - (float)hi);
            }
            int pb = permk(c4);               // group of 4 stays contiguous
            *(uint2*)&sAh[r][pb] = ph.u;      // ds_store_b64
            *(uint2*)&sAl[r][pb] = pl.u;
        }
        // ---- stage B tile (32 x 64 fp32 -> bf16 hi/lo, [n][perm(k)]) ------
#pragma unroll
        for (int it = 0; it < 2; ++it) {
            int p  = tid + it * 256;
            int r  = p >> 4;                  // k local: 0..31
            int c4 = (p & 15) * 4;            // n local: 0..60
            const float* bp = Wc + (size_t)(kk + r) * FDIM + col0 + c4;
            float4 v4 = *(const float4*)bp;
            if (kk + 32 < FDIM) __builtin_prefetch(bp + 32 * FDIM, 0, 1);
            float f[4] = {v4.x, v4.y, v4.z, v4.w};
            int pk = permk(r);
#pragma unroll
            for (int j = 0; j < 4; ++j) {
                __bf16 hi = (__bf16)f[j];
                sBh[c4 + j][pk] = hi;
                sBl[c4 + j][pk] = (__bf16)(f[j] - (float)hi);
            }
        }
        __syncthreads();

        // ---- fragments: 2 x ds_load_b128 each, no packing -----------------
        FragU ah, al, bh0, bl0, bh1, bl1;
        {
            const uint4* pa = (const uint4*)&sAh[m0 + lm][hb * 16];
            ah.q[0] = pa[0]; ah.q[1] = pa[1];
            const uint4* pb = (const uint4*)&sAl[m0 + lm][hb * 16];
            al.q[0] = pb[0]; al.q[1] = pb[1];
            const uint4* p0h = (const uint4*)&sBh[n0 + lm][hb * 16];
            bh0.q[0] = p0h[0]; bh0.q[1] = p0h[1];
            const uint4* p0l = (const uint4*)&sBl[n0 + lm][hb * 16];
            bl0.q[0] = p0l[0]; bl0.q[1] = p0l[1];
            const uint4* p1h = (const uint4*)&sBh[n0 + 16 + lm][hb * 16];
            bh1.q[0] = p1h[0]; bh1.q[1] = p1h[1];
            const uint4* p1l = (const uint4*)&sBl[n0 + 16 + lm][hb * 16];
            bl1.q[0] = p1l[0]; bl1.q[1] = p1l[1];
        }

        // ---- bf16-split fp32 GEMM: 6 WMMAs per K slab ---------------------
        acc0 = __builtin_amdgcn_wmma_f32_16x16x32_bf16(false, ah.v, false, bh0.v,
                                                       (short)0, acc0, false, false);
        acc0 = __builtin_amdgcn_wmma_f32_16x16x32_bf16(false, al.v, false, bh0.v,
                                                       (short)0, acc0, false, false);
        acc0 = __builtin_amdgcn_wmma_f32_16x16x32_bf16(false, ah.v, false, bl0.v,
                                                       (short)0, acc0, false, false);
        acc1 = __builtin_amdgcn_wmma_f32_16x16x32_bf16(false, ah.v, false, bh1.v,
                                                       (short)0, acc1, false, false);
        acc1 = __builtin_amdgcn_wmma_f32_16x16x32_bf16(false, al.v, false, bh1.v,
                                                       (short)0, acc1, false, false);
        acc1 = __builtin_amdgcn_wmma_f32_16x16x32_bf16(false, ah.v, false, bl1.v,
                                                       (short)0, acc1, false, false);
        __syncthreads();
    }

    // C/D layout (ISA 7.12.2): VGPR r holds row M = r + 8*hb, lane lm = N.
#pragma unroll
    for (int r = 0; r < 8; ++r) {
        int grow = row0 + m0 + r + 8 * hb;
        if (grow < NNODES) {
            hc[(size_t)grow * FDIM + col0 + n0 + lm]      = acc0[r];
            hc[(size_t)grow * FDIM + col0 + n0 + 16 + lm] = acc1[r];
        }
    }
}

// ---------------------------------------------------- bias + self loop ------
__global__ __launch_bounds__(256) void k_init_out(const float* __restrict__ h,
                                                  const float* __restrict__ dinv,
                                                  const float* __restrict__ bb,
                                                  float* __restrict__ out) {
    int idx = blockIdx.x * 256 + threadIdx.x;
    const int total = CH * NNODES * FDIM;
    if (idx < total) {
        int f  = idx & (FDIM - 1);
        int n  = (idx / FDIM) % NNODES;
        int cc = idx / (FDIM * NNODES);
        float di = dinv[n];
        out[idx] = bb[cc * FDIM + f] + di * di * h[idx];
    }
}

// --------------------------------------------------------------- scatter ----
// One block per edge; 256 threads = one full feature row; loop channels.
__global__ __launch_bounds__(256) void k_scatter(const int* __restrict__ ei,
                                                 const float* __restrict__ nrm,
                                                 const float* __restrict__ h,
                                                 float* __restrict__ out) {
    int e   = blockIdx.x;
    int row = ei[e];
    int col = ei[EDGES + e];
    float w = nrm[e];
    int f   = threadIdx.x;
#pragma unroll
    for (int c = 0; c < CH; ++c) {
        float v = w * h[((size_t)c * NNODES + row) * FDIM + f];
        atomicAdd(&out[((size_t)c * NNODES + col) * FDIM + f], v);
    }
}

// ---------------------------------------------------------------------------
extern "C" void kernel_launch(void* const* d_in, const int* in_sizes, int n_in,
                              void* d_out, int out_size, void* d_ws, size_t ws_size,
                              hipStream_t stream) {
    XPtrs xp;
    for (int i = 0; i < CH; ++i) xp.p[i] = (const float*)d_in[i];
    const int*   ei = (const int*)d_in[10];
    const float* ew = (const float*)d_in[11];
    const float* Wm = (const float*)d_in[12];
    const float* bb = (const float*)d_in[13];
    float* out = (float*)d_out;

    // workspace: deg[N] | dinv[N] | norm[E] | h[C*N*F]  (~103 MB)
    float* deg  = (float*)d_ws;
    float* dinv = deg + NNODES;
    float* nrm  = dinv + NNODES;
    float* h    = nrm + EDGES;

    k_init_deg<<<(NNODES + 255) / 256, 256, 0, stream>>>(deg);
    k_deg<<<(EDGES + 255) / 256, 256, 0, stream>>>(ei, ew, deg);
    k_dinv<<<(NNODES + 255) / 256, 256, 0, stream>>>(deg, dinv);
    k_norm<<<(EDGES + 255) / 256, 256, 0, stream>>>(ei, ew, dinv, nrm);

    dim3 g((NNODES + 63) / 64, FDIM / 64, CH);
    k_gemm<<<g, 256, 0, stream>>>(xp, Wm, h);

    const int total = CH * NNODES * FDIM;
    k_init_out<<<(total + 255) / 256, 256, 0, stream>>>(h, dinv, bb, out);
    k_scatter<<<EDGES, 256, 0, stream>>>(ei, nrm, h, out);
}
